// RBFBias_20555713478724
// MI455X (gfx1250) — matrix operand order repaired and checked by reading, not compiled, not run on hardware
//
#include <hip/hip_runtime.h>

typedef __attribute__((ext_vector_type(16))) _Float16 v16h;
typedef __attribute__((ext_vector_type(8)))  float    v8f;

#define B_ 2
#define H_ 8
#define S_ 2048
#define D_ 64
#define SCALE 0.125f   // 1/sqrt(64)

// 16-bit A-matrix (16x32) K index for half h (0..15) in lane-group g (contiguous: k = 16*g + h, matches B fragments)
__device__ __forceinline__ int amapK(int h, int g) {
    return h + 16 * g;
}

__global__ __launch_bounds__(128, 1) void rbf_flash_attn(
    const float* __restrict__ qg, const float* __restrict__ kg,
    const float* __restrict__ vg, const float* __restrict__ qsg,
    const float* __restrict__ ksg, const float* __restrict__ alphag,
    const float* __restrict__ betag, float* __restrict__ outg)
{
    // LDS staging (strides chosen so v16h fragment loads are 32B-aligned)
    __shared__ _Float16 Ksh[32][80];      // key chunk, row-major      (5120 B)
    __shared__ _Float16 Vts[64][48];      // value chunk, transposed   (6144 B)
    __shared__ _Float16 Psh[4][16][32];   // per-wave P C->A relayout  (4096 B)
    __shared__ float    KPs[32][3];       // key position features     ( 384 B)

    const int tid  = threadIdx.x;
    const int wave = tid >> 5;
    const int lane = tid & 31;
    const int g    = lane >> 4;       // lane half-group
    const int n    = lane & 15;       // B/C/D column within 16
    const int m    = lane & 15;       // A row

    const int bid  = blockIdx.x;
    const int qblk = bid & 31;        // S/64 = 32 query blocks
    const int bh   = bid >> 5;
    const int hh   = bh & 7;          // head
    const int b    = bh >> 3;         // batch

    const int q0 = qblk * 64 + wave * 16;

    const float alpha_h = alphag[hh];
    const float beta_h  = betag[hh];
    const size_t bh_off = ((size_t)(b * H_ + hh)) * S_;

    // ---- Q tile -> two A fragments (D=64 split into 2 x K=32), fp32->f16 ----
    union { v16h v; _Float16 e[16]; } aq[2];
    {
        const float* qrow = qg + (bh_off + q0 + m) * D_;
#pragma unroll
        for (int c = 0; c < 2; ++c)
#pragma unroll
            for (int h2 = 0; h2 < 16; ++h2)
                aq[c].e[h2] = (_Float16)qrow[c * 32 + amapK(h2, g)];
    }

    // ---- query position features for this lane's 8 C-layout rows ----
    float qp[8][3];
#pragma unroll
    for (int r = 0; r < 8; ++r) {
        const float* p = qsg + ((size_t)b * S_ + q0 + 8 * g + r) * 3;
        qp[r][0] = p[0]; qp[r][1] = p[1]; qp[r][2] = p[2];
    }

    v8f o[4];
#pragma unroll
    for (int f = 0; f < 4; ++f) o[f] = (v8f){};
    float mrow[8], lrow[8];
#pragma unroll
    for (int r = 0; r < 8; ++r) { mrow[r] = -INFINITY; lrow[r] = 0.0f; }

    // ---- software pipeline: prefetch chunk 0 into registers ----
    float kreg[16], vreg[16], ksreg = 0.0f;
    {
        const float* kp_ = kg + bh_off * D_;
        const float* vp_ = vg + bh_off * D_;
#pragma unroll
        for (int i = 0; i < 16; ++i) {
            kreg[i] = kp_[tid + 128 * i];
            vreg[i] = vp_[tid + 128 * i];
        }
        if (tid < 96) ksreg = ksg[(size_t)b * S_ * 3 + tid];
    }

    for (int kc = 0; kc < S_ / 32; ++kc) {
        // ---- convert + store the prefetched chunk: K row-major, V transposed ----
#pragma unroll
        for (int i = 0; i < 16; ++i) {
            int idx = tid + 128 * i;            // 0..2047
            int row = idx >> 6, col = idx & 63;
            Ksh[row][col] = (_Float16)kreg[i];
            Vts[col][row] = (_Float16)vreg[i];
        }
        if (tid < 96) (&KPs[0][0])[tid] = ksreg;
        __syncthreads();

        // ---- issue next chunk's global loads; they drain during compute ----
        if (kc + 1 < S_ / 32) {
            const int k1 = (kc + 1) * 32;
            const float* kp_ = kg + (bh_off + k1) * D_;
            const float* vp_ = vg + (bh_off + k1) * D_;
#pragma unroll
            for (int i = 0; i < 16; ++i) {
                kreg[i] = kp_[tid + 128 * i];
                vreg[i] = vp_[tid + 128 * i];
            }
            if (tid < 96) ksreg = ksg[((size_t)b * S_ + k1) * 3 + tid];
        }

        // key position features for this lane's two columns
        float kp0[3], kp1[3];
#pragma unroll
        for (int j = 0; j < 3; ++j) { kp0[j] = KPs[n][j]; kp1[j] = KPs[16 + n][j]; }

        // ---- S = Q * K^T : 2 N-tiles x 2 K-chunks = 4 WMMAs ----
        v8f sa[2];
#pragma unroll
        for (int t = 0; t < 2; ++t) {
            v8f acc = (v8f){};
#pragma unroll
            for (int c = 0; c < 2; ++c) {
                const v16h bk = *(const v16h*)&Ksh[t * 16 + n][c * 32 + 16 * g];
                acc = __builtin_amdgcn_wmma_f32_16x16x32_f16(
                        false, aq[c].v, false, bk, (short)0, acc, false, false);
            }
            sa[t] = acc;
        }

        // ---- RBF bias + online softmax (row reductions in 16-lane groups) ----
#pragma unroll
        for (int r = 0; r < 8; ++r) {
            float d0x = qp[r][0]-kp0[0], d0y = qp[r][1]-kp0[1], d0z = qp[r][2]-kp0[2];
            float d1x = qp[r][0]-kp1[0], d1y = qp[r][1]-kp1[1], d1z = qp[r][2]-kp1[2];
            float s0 = sa[0][r] * SCALE +
                       alpha_h * __expf(-beta_h * (d0x*d0x + d0y*d0y + d0z*d0z));
            float s1 = sa[1][r] * SCALE +
                       alpha_h * __expf(-beta_h * (d1x*d1x + d1y*d1y + d1z*d1z));
            float vmax = fmaxf(s0, s1);
#pragma unroll
            for (int off = 1; off <= 8; off <<= 1)
                vmax = fmaxf(vmax, __shfl_xor(vmax, off, 32));
            float mn   = fmaxf(mrow[r], vmax);
            float resc = __expf(mrow[r] - mn);
            float p0 = __expf(s0 - mn), p1 = __expf(s1 - mn);
            float ps = p0 + p1;
#pragma unroll
            for (int off = 1; off <= 8; off <<= 1)
                ps += __shfl_xor(ps, off, 32);
            lrow[r] = lrow[r] * resc + ps;
            mrow[r] = mn;
#pragma unroll
            for (int f = 0; f < 4; ++f) o[f][r] *= resc;
            Psh[wave][r + 8 * g][n]      = (_Float16)p0;
            Psh[wave][r + 8 * g][16 + n] = (_Float16)p1;
        }

        // ---- P in A layout (wave-private LDS round trip; DS in-order per wave,
        //      compiler tracks the alias through DScnt) ----
        union { v16h v; _Float16 e[16]; } ap;
#pragma unroll
        for (int h2 = 0; h2 < 16; ++h2)
            ap.e[h2] = Psh[wave][m][amapK(h2, g)];

        // ---- O += P * V : 4 N-tiles = 4 WMMAs ----
#pragma unroll
        for (int f = 0; f < 4; ++f) {
            const v16h bv = *(const v16h*)&Vts[f * 16 + n][16 * g];
            o[f] = __builtin_amdgcn_wmma_f32_16x16x32_f16(
                       false, ap.v, false, bv, (short)0, o[f], false, false);
        }
        __syncthreads();
    }

    // ---- normalize + store ----
#pragma unroll
    for (int r = 0; r < 8; ++r) {
        float inv = 1.0f / lrow[r];
        float* orow = outg + (bh_off + q0 + r + 8 * g) * D_;
#pragma unroll
        for (int f = 0; f < 4; ++f)
            orow[f * 16 + n] = o[f][r] * inv;
    }
}

extern "C" void kernel_launch(void* const* d_in, const int* in_sizes, int n_in,
                              void* d_out, int out_size, void* d_ws, size_t ws_size,
                              hipStream_t stream) {
    (void)in_sizes; (void)n_in; (void)out_size; (void)d_ws; (void)ws_size;
    const float* q     = (const float*)d_in[0];
    const float* k     = (const float*)d_in[1];
    const float* v     = (const float*)d_in[2];
    const float* qs_s  = (const float*)d_in[3];
    const float* ks_s  = (const float*)d_in[4];
    const float* alpha = (const float*)d_in[5];
    const float* beta  = (const float*)d_in[6];
    float* out = (float*)d_out;

    dim3 grid(B_ * H_ * (S_ / 64));   // 512 blocks
    dim3 block(128);                  // 4 waves; 64 query rows / block
    rbf_flash_attn<<<grid, block, 0, stream>>>(q, k, v, qs_s, ks_s, alpha, beta, out);
}
